// MaskcomputeMoE_56547539419489
// MI455X (gfx1250) — matrix operand order repaired and checked by reading, not compiled, not run on hardware
//
#include <hip/hip_runtime.h>
#include <hip/hip_bf16.h>
#include <math.h>

typedef _Float16 v16h __attribute__((ext_vector_type(16)));
typedef _Float16 v8h  __attribute__((ext_vector_type(8)));
typedef float    v8f  __attribute__((ext_vector_type(8)));
typedef float    v4f  __attribute__((ext_vector_type(4)));

#define Bb 32
#define Ll 1024
#define Cc 1024
#define Ee 64
#define Tt 8

// ---------------------------------------------------------------------------
// Kernel A: hgate[b,c] = gelu(x2[b,:] @ W1[:,c] + b1[c]) -> f16
//           W2T[n,k]   = W2[k,n]  (first 64 cols only)   -> f16
// ---------------------------------------------------------------------------
__global__ __launch_bounds__(256) void kA(const float* __restrict__ x2,
                                          const float* __restrict__ W1,
                                          const float* __restrict__ b1,
                                          const float* __restrict__ W2,
                                          _Float16* __restrict__ hgate,
                                          _Float16* __restrict__ w2t) {
  int idx = blockIdx.x * 256 + threadIdx.x;
  if (idx < Bb * Cc) {
    int b = idx >> 10, c = idx & (Cc - 1);
    float v = b1[c];
#pragma unroll
    for (int t = 0; t < Tt; ++t) v += x2[b * Tt + t] * W1[t * Cc + c];
    // exact gelu: x * 0.5 * (1 + erf(x/sqrt(2)))
    float g = 0.5f * v * (1.0f + erff(v * 0.70710678118654752f));
    hgate[idx] = (_Float16)g;
  } else {
    int j = idx - Bb * Cc;          // 0..65535
    if (j < Ee * Cc) {
      int n = j >> 10, k = j & (Cc - 1);
      w2t[n * Cc + k] = (_Float16)W2[k * (2 * Ee) + n];
    }
  }
}

// ---------------------------------------------------------------------------
// Kernel B: logits = hgate @ W2T^T (32x64, K=1024) via v_wmma_f32_16x16x32_f16
//           then softmax(+1e-4), top-2, expert_mask, per-batch params.
// One workgroup, 8 waves; wave w owns output tile (mt = w>>2, nt = w&3).
// ---------------------------------------------------------------------------
__global__ __launch_bounds__(256) void kB(const _Float16* __restrict__ hgate,
                                          const _Float16* __restrict__ w2t,
                                          float* __restrict__ emask_out,
                                          int* __restrict__ pe,
                                          float* __restrict__ pg) {
  __shared__ float slog[Bb * Ee];   // logits -> probs, in place
  __shared__ float sth[Bb];         // per-batch threshold (2nd-largest prob)

  const int tid  = threadIdx.x;
  const int wave = tid >> 5;
  const int lane = tid & 31;
  const int mt = wave >> 2, nt = wave & 3;
  const int m  = lane & 15;
  const int hi = lane >> 4;         // K half-select per ISA layout
  const int n  = lane & 15;

  const _Float16* arow = hgate + (mt * 16 + m) * Cc;
  const _Float16* brow = w2t   + (nt * 16 + n) * Cc;

  v8f acc = {};
#pragma unroll 4
  for (int kb = 0; kb < Cc / 32; ++kb) {
    const int k0 = kb * 32;
    // A 16x32 f16 layout: a[h] = A[m][ (h<8 ? h : h+8) + hi*8 + k0 ]
    v8h alo = *(const v8h*)(arow + k0 + hi * 8);        // K = k0+hi*8   .. +7
    v8h ahi = *(const v8h*)(arow + k0 + 16 + hi * 8);   // K = k0+16+hi*8.. +7
    v16h a = __builtin_shufflevector(alo, ahi, 0, 1, 2, 3, 4, 5, 6, 7,
                                     8, 9, 10, 11, 12, 13, 14, 15);
    // B 32x16 f16 layout: b[h] = B[ (h<8 ? h : h+8) + hi*8 + k0 ][n]
    v8h blo = *(const v8h*)(brow + k0 + hi * 8);
    v8h bhi = *(const v8h*)(brow + k0 + 16 + hi * 8);
    v16h b = __builtin_shufflevector(blo, bhi, 0, 1, 2, 3, 4, 5, 6, 7,
                                     8, 9, 10, 11, 12, 13, 14, 15);
    acc = __builtin_amdgcn_wmma_f32_16x16x32_f16(false, a, false, b,
                                                 (short)0, acc, false, false);
  }
  // C/D layout: acc[j] = D[mt*16 + j + hi*8][nt*16 + n]
#pragma unroll
  for (int j = 0; j < 8; ++j)
    slog[(mt * 16 + hi * 8 + j) * Ee + nt * 16 + n] = acc[j];
  __syncthreads();

  if (tid < Bb) {
    float* row = slog + tid * Ee;
    float mx = row[0];
    for (int e = 1; e < Ee; ++e) mx = fmaxf(mx, row[e]);
    float sum = 0.0f;
    for (int e = 0; e < Ee; ++e) { float v = __expf(row[e] - mx); row[e] = v; sum += v; }
    float inv = 1.0f / sum;
    for (int e = 0; e < Ee; ++e) row[e] = row[e] * inv + 1.0e-4f;
    // top-2, first-occurrence tie-breaking like jax top_k
    int e1 = 0; float g1 = row[0];
    for (int e = 1; e < Ee; ++e) if (row[e] > g1) { g1 = row[e]; e1 = e; }
    int e2 = (e1 == 0) ? 1 : 0; float g2 = row[e2];
    for (int e = 0; e < Ee; ++e) if (e != e1 && row[e] > g2) { g2 = row[e]; e2 = e; }
    sth[tid] = g2;
    pe[tid] = e1; pe[Bb + tid] = e2;
    pg[tid] = g1; pg[Bb + tid] = g2;
  }
  __syncthreads();

  if (tid < Ee) {  // expert_mask[e] = sum_b probs[b,e] * (probs[b,e] >= thresh_b)
    float s = 0.0f;
    for (int b = 0; b < Bb; ++b) {
      float v = slog[b * Ee + tid];
      if (v >= sth[b]) s += v;
    }
    emask_out[tid] = s;
  }
}

// ---------------------------------------------------------------------------
// Kernel C: the bandwidth kernel. One block per (b,l) row; 256 threads x
// float4 = 1024 floats. y = x * (g1*G[b,e1,l] + g2*G[b,e2,l]).
// Non-temporal: x and y are each touched exactly once.
// ---------------------------------------------------------------------------
__global__ __launch_bounds__(256) void kC(const float* __restrict__ x,
                                          const float* __restrict__ G,
                                          const int* __restrict__ pe,
                                          const float* __restrict__ pg,
                                          float* __restrict__ y) {
  const int row = blockIdx.x;           // 0..32767
  const int b = row >> 10;
  const int l = row & (Ll - 1);
  const int e1 = pe[b], e2 = pe[Bb + b];
  const float g1 = pg[b], g2 = pg[Bb + b];
  const float s = g1 * G[((size_t)b * Ee + e1) * Ll + l]
                + g2 * G[((size_t)b * Ee + e2) * Ll + l];

  const v4f* __restrict__ xin = (const v4f*)(x + (size_t)row * Cc);
  v4f* __restrict__ yo = (v4f*)(y + (size_t)row * Cc);
  v4f v = __builtin_nontemporal_load(xin + threadIdx.x);
  v *= s;
  __builtin_nontemporal_store(v, yo + threadIdx.x);
}

// ---------------------------------------------------------------------------
// Kernel D: token_mask[l] = sum_b (G[b,e1,l] + G[b,e2,l])  (raw G, no gates)
// ---------------------------------------------------------------------------
__global__ __launch_bounds__(256) void kD(const float* __restrict__ G,
                                          const int* __restrict__ pe,
                                          float* __restrict__ tmask) {
  const int l = blockIdx.x * 256 + threadIdx.x;   // 0..1023
  float s = 0.0f;
#pragma unroll 4
  for (int b = 0; b < Bb; ++b)
    s += G[((size_t)b * Ee + pe[b]) * Ll + l]
       + G[((size_t)b * Ee + pe[Bb + b]) * Ll + l];
  tmask[l] = s;
}

// ---------------------------------------------------------------------------
extern "C" void kernel_launch(void* const* d_in, const int* in_sizes, int n_in,
                              void* d_out, int out_size, void* d_ws, size_t ws_size,
                              hipStream_t stream) {
  const float* x  = (const float*)d_in[0];  // (32,1024,1024)
  const float* x2 = (const float*)d_in[1];  // (32,8)
  const float* G  = (const float*)d_in[2];  // (32,64,1024)
  const float* W1 = (const float*)d_in[3];  // (8,1024)
  const float* b1 = (const float*)d_in[4];  // (1024,)
  const float* W2 = (const float*)d_in[5];  // (1024,128)
  // d_in[6] = k (==2, hardcoded)

  float* y     = (float*)d_out;                       // 33554432
  float* emask = y + (size_t)Bb * Ll * Cc;            // +64
  float* tmask = emask + Ee;                          // +1024

  char* ws = (char*)d_ws;
  _Float16* hgate = (_Float16*)(ws);                  //  64 KB
  _Float16* w2t   = (_Float16*)(ws + 65536);          // 128 KB
  int*      pe    = (int*)  (ws + 196608);            // 256 B (e1[32], e2[32])
  float*    pg    = (float*)(ws + 196864);            // 256 B (g1[32], g2[32])

  kA<<<(Bb * Cc + Ee * Cc) / 256, 256, 0, stream>>>(x2, W1, b1, W2, hgate, w2t);
  kB<<<1, 256, 0, stream>>>(hgate, w2t, emask, pe, pg);
  kC<<<Bb * Ll, 256, 0, stream>>>(x, G, pe, pg, y);
  kD<<<Ll / 256, 256, 0, stream>>>(G, pe, tmask);
}
